// KNNRouter_52347061403861
// MI455X (gfx1250) — matrix operand order repaired and checked by reading, not compiled, not run on hardware
//
#include <hip/hip_runtime.h>
#include <hip/hip_bf16.h>

typedef __attribute__((ext_vector_type(16))) __bf16 bf16x16;
typedef __attribute__((ext_vector_type(8)))  float  floatx8;
typedef unsigned int u32;
typedef unsigned short u16;

#define D_DIM 1024
#define TOPK 8
#define M_BLK 32                 // query rows per block (2 row-tiles / wave)
#define V_SPLIT 2                // vocab halves per row-block
#define NEG_INF (-3.0e38f)

__device__ __forceinline__ u32 bf16_rne(float f) {
    u32 u = __float_as_uint(f);
    return (u + 0x7fffu + ((u >> 16) & 1u)) >> 16;
}
__device__ __forceinline__ u32 pack2(float lo, float hi) {
    return bf16_rne(lo) | (bf16_rne(hi) << 16);
}

// ---------------------------------------------------------------------------
// Zero-fill the (N,V) probs region with wide stores.
// ---------------------------------------------------------------------------
__global__ void zero_kernel(float4* __restrict__ p, long long n4) {
    long long i = (long long)blockIdx.x * blockDim.x + threadIdx.x;
    long long stride = (long long)gridDim.x * blockDim.x;
    float4 z = {0.f, 0.f, 0.f, 0.f};
    for (; i < n4; i += stride) p[i] = z;
}

// ---------------------------------------------------------------------------
// L2-normalize vocab rows (f32 -> bf16).  One wave per row, 8 rows per block.
// ---------------------------------------------------------------------------
__global__ void vocab_norm_kernel(const float* __restrict__ vocab,
                                  u16* __restrict__ vn, int V) {
    int row  = blockIdx.x * 8 + (threadIdx.x >> 5);
    int lane = threadIdx.x & 31;
    if (row >= V) return;
    const float4* src = (const float4*)(vocab + (size_t)row * D_DIM);
    float4 r[8];
    float ss = 0.f;
#pragma unroll
    for (int i = 0; i < 8; ++i) {
        r[i] = src[lane + 32 * i];
        ss += r[i].x * r[i].x + r[i].y * r[i].y + r[i].z * r[i].z + r[i].w * r[i].w;
    }
#pragma unroll
    for (int off = 16; off > 0; off >>= 1) ss += __shfl_xor(ss, off, 32);
    float sc = 1.f / fmaxf(sqrtf(ss), 1e-12f);
    uint2* dst = (uint2*)(vn + (size_t)row * D_DIM);
#pragma unroll
    for (int i = 0; i < 8; ++i) {
        uint2 o;
        o.x = pack2(r[i].x * sc, r[i].y * sc);
        o.y = pack2(r[i].z * sc, r[i].w * sc);
        dst[lane + 32 * i] = o;
    }
}

// ---------------------------------------------------------------------------
// Fused: normalize 32 query rows -> bf16 WMMA GEMM (2 row-tiles share each B
// fragment) over one V-half -> streaming per-row top-8 -> candidate dump.
// grid.x = (N/M_BLK) * V_SPLIT ; blockDim = 256 (8 waves).
// ---------------------------------------------------------------------------
__global__ void __launch_bounds__(256, 1)
knn_router_kernel(const float* __restrict__ emb,
                  const u16* __restrict__ vn,
                  float* __restrict__ candS,   // [N][V_SPLIT*TOPK]
                  int*   __restrict__ candI,
                  int V) {
    __shared__ u16   sQ[M_BLK * D_DIM];      // 64 KB: bf16 query tile
    __shared__ float sC[8][M_BLK * 16];      // 16 KB: per-wave 32x16 sim tile
    __shared__ float sTS[8][M_BLK][TOPK];    //  8 KB
    __shared__ int   sTI[8][M_BLK][TOPK];    //  8 KB

    const int tid      = threadIdx.x;
    const int w        = tid >> 5;
    const int lane     = tid & 31;
    const int l15      = lane & 15;
    const int halfsel  = lane >> 4;          // 0/1 : K-half selector
    const int rowBlock = blockIdx.x >> 1;
    const int split    = blockIdx.x & 1;
    const int rowBase  = rowBlock * M_BLK;
    const int vHalf    = V / V_SPLIT;
    const int vBeg     = split * vHalf;
    const int vEnd     = vBeg + vHalf;

    // ---- phase 0: normalize this block's 32 query rows into LDS (bf16) ----
    for (int rr = w; rr < M_BLK; rr += 8) {
        const float4* src = (const float4*)(emb + (size_t)(rowBase + rr) * D_DIM);
        float4 r[8];
        float ss = 0.f;
#pragma unroll
        for (int i = 0; i < 8; ++i) {
            r[i] = src[lane + 32 * i];
            ss += r[i].x * r[i].x + r[i].y * r[i].y + r[i].z * r[i].z + r[i].w * r[i].w;
        }
#pragma unroll
        for (int off = 16; off > 0; off >>= 1) ss += __shfl_xor(ss, off, 32);
        float sc = 1.f / fmaxf(sqrtf(ss), 1e-12f);
        uint2* dst = ((uint2*)sQ) + rr * 256;
#pragma unroll
        for (int i = 0; i < 8; ++i) {
            uint2 o;
            o.x = pack2(r[i].x * sc, r[i].y * sc);
            o.y = pack2(r[i].z * sc, r[i].w * sc);
            dst[lane + 32 * i] = o;
        }
    }
    __syncthreads();

    // ---- streaming top-8 state: lane owns query row `lane` (0..31) ----
    float ts[TOPK]; int ti[TOPK];
#pragma unroll
    for (int j = 0; j < TOPK; ++j) { ts[j] = NEG_INF; ti[j] = 0; }
    float tmin = NEG_INF; int tminpos = 0;

    const u16* sQrow0 = sQ + l15 * D_DIM;           // A row-tile 0
    const u16* sQrow1 = sQ + (16 + l15) * D_DIM;    // A row-tile 1

    // ---- main loop: wave w owns cols {vBeg + w*16 + 128*t} ----
    for (int v0 = vBeg + w * 16; v0 < vEnd; v0 += 128) {
        const u16* vrow = vn + (size_t)(v0 + l15) * D_DIM;
        if (v0 + 128 < vEnd)
            __builtin_prefetch(vn + (size_t)(v0 + 128 + l15) * D_DIM, 0, 1);

        floatx8 c0 = {0.f, 0.f, 0.f, 0.f, 0.f, 0.f, 0.f, 0.f};
        floatx8 c1 = {0.f, 0.f, 0.f, 0.f, 0.f, 0.f, 0.f, 0.f};
#pragma unroll 4
        for (int k0 = 0; k0 < D_DIM; k0 += 32) {
            union { bf16x16 v; uint4 q[2]; } a0, a1, b;
            // B (32x16 bf16): lanes 0-15 hold K = k0..k0+7 & k0+16..k0+23, half-wave +8
            const int kb = k0 + halfsel * 8;
            b.q[0] = *(const uint4*)(vrow + kb);
            b.q[1] = *(const uint4*)(vrow + kb + 16);
            // A (16x32 bf16): lanes 0-15 hold K = k0..k0+7 & k0+16..k0+23
            const int ka = k0 + halfsel * 8;
            a0.q[0] = *(const uint4*)(sQrow0 + ka);
            a0.q[1] = *(const uint4*)(sQrow0 + ka + 16);
            a1.q[0] = *(const uint4*)(sQrow1 + ka);
            a1.q[1] = *(const uint4*)(sQrow1 + ka + 16);
            c0 = __builtin_amdgcn_wmma_f32_16x16x32_bf16(
                     false, a0.v, false, b.v, (short)0, c0, false, false);
            c1 = __builtin_amdgcn_wmma_f32_16x16x32_bf16(
                     false, a1.v, false, b.v, (short)0, c1, false, false);
        }

        // spill 32x16 tile to per-wave LDS patch
#pragma unroll
        for (int j = 0; j < 8; ++j) {
            sC[w][(j + halfsel * 8) * 16 + l15]        = c0[j];
            sC[w][(16 + j + halfsel * 8) * 16 + l15]   = c1[j];
        }
        __builtin_amdgcn_wave_barrier();

        // lane scans its row's 16 new sims (4x b128 from LDS)
        const float4* rowv = (const float4*)(sC[w] + lane * 16);
#pragma unroll
        for (int q4 = 0; q4 < 4; ++q4) {
            float4 vv = rowv[q4];
            float vals[4] = {vv.x, vv.y, vv.z, vv.w};
#pragma unroll
            for (int e = 0; e < 4; ++e) {
                float v = vals[e];
                if (v > tmin) {                         // rare after warm-up
                    int col = v0 + q4 * 4 + e;
#pragma unroll
                    for (int j = 0; j < TOPK; ++j)
                        if (j == tminpos) { ts[j] = v; ti[j] = col; }
                    tmin = ts[0]; tminpos = 0;
#pragma unroll
                    for (int j = 1; j < TOPK; ++j)
                        if (ts[j] < tmin) { tmin = ts[j]; tminpos = j; }
                }
            }
        }
        __builtin_amdgcn_wave_barrier();
    }

    // ---- publish per-wave top-8, merge across waves in wave 0 ----
#pragma unroll
    for (int j = 0; j < TOPK; ++j) {
        sTS[w][lane][j] = ts[j];
        sTI[w][lane][j] = ti[j];
    }
    __syncthreads();

    if (w == 0) {   // all 32 lanes: lane = row 0..31
        float fs[TOPK]; int fi[TOPK];
#pragma unroll
        for (int j = 0; j < TOPK; ++j) { fs[j] = NEG_INF; fi[j] = 0; }
        float fmin = NEG_INF; int fminpos = 0;
        for (int ww = 0; ww < 8; ++ww) {
#pragma unroll
            for (int j = 0; j < TOPK; ++j) {
                float v = sTS[ww][lane][j];
                int  id = sTI[ww][lane][j];
                if (v > fmin) {
#pragma unroll
                    for (int q = 0; q < TOPK; ++q)
                        if (q == fminpos) { fs[q] = v; fi[q] = id; }
                    fmin = fs[0]; fminpos = 0;
#pragma unroll
                    for (int q = 1; q < TOPK; ++q)
                        if (fs[q] < fmin) { fmin = fs[q]; fminpos = q; }
                }
            }
        }
        size_t rowg = (size_t)(rowBase + lane);
        float* cs = candS + rowg * (V_SPLIT * TOPK) + split * TOPK;
        int*   ci = candI + rowg * (V_SPLIT * TOPK) + split * TOPK;
#pragma unroll
        for (int j = 0; j < TOPK; ++j) { cs[j] = fs[j]; ci[j] = fi[j]; }
    }
}

// ---------------------------------------------------------------------------
// Final merge: per row combine V_SPLIT*TOPK candidates -> top-8 -> softmax
// -> scatter into probs + hard token id.  One thread per row.
// ---------------------------------------------------------------------------
__global__ void merge_kernel(const float* __restrict__ candS,
                             const int* __restrict__ candI,
                             float* __restrict__ probs,
                             float* __restrict__ toks,
                             int N, int V) {
    int row = blockIdx.x * blockDim.x + threadIdx.x;
    if (row >= N) return;
    const float* cs = candS + (size_t)row * (V_SPLIT * TOPK);
    const int*   ci = candI + (size_t)row * (V_SPLIT * TOPK);

    float fs[TOPK]; int fi[TOPK];
#pragma unroll
    for (int j = 0; j < TOPK; ++j) { fs[j] = NEG_INF; fi[j] = 0; }
    float fmin = NEG_INF; int fminpos = 0;
#pragma unroll
    for (int j = 0; j < V_SPLIT * TOPK; ++j) {
        float v = cs[j];
        int  id = ci[j];
        if (v > fmin) {
#pragma unroll
            for (int q = 0; q < TOPK; ++q)
                if (q == fminpos) { fs[q] = v; fi[q] = id; }
            fmin = fs[0]; fminpos = 0;
#pragma unroll
            for (int q = 1; q < TOPK; ++q)
                if (fs[q] < fmin) { fmin = fs[q]; fminpos = q; }
        }
    }
    float m = fs[0]; int am = fi[0];
#pragma unroll
    for (int j = 1; j < TOPK; ++j)
        if (fs[j] > m) { m = fs[j]; am = fi[j]; }
    float p[TOPK]; float sum = 0.f;
#pragma unroll
    for (int j = 0; j < TOPK; ++j) { p[j] = __expf(fs[j] - m); sum += p[j]; }
    float inv = 1.f / sum;
#pragma unroll
    for (int j = 0; j < TOPK; ++j)
        probs[(size_t)row * V + fi[j]] = p[j] * inv;
    toks[row] = (float)am;
}

// ---------------------------------------------------------------------------
extern "C" void kernel_launch(void* const* d_in, const int* in_sizes, int n_in,
                              void* d_out, int out_size, void* d_ws, size_t ws_size,
                              hipStream_t stream) {
    const float* emb   = (const float*)d_in[0];
    const float* vocab = (const float*)d_in[1];
    const int N = in_sizes[0] / D_DIM;     // 4096
    const int V = in_sizes[1] / D_DIM;     // 32000

    float* probs = (float*)d_out;
    float* toks  = probs + (size_t)N * V;

    // workspace layout: [bf16 vocab 64MB][candS 256KB][candI 256KB]
    u16*   vn    = (u16*)d_ws;
    size_t vnB   = (size_t)V * D_DIM * sizeof(u16);
    float* candS = (float*)((char*)d_ws + vnB);
    int*   candI = (int*)((char*)d_ws + vnB + (size_t)N * V_SPLIT * TOPK * sizeof(float));

    long long n4 = ((long long)N * V) / 4;
    zero_kernel<<<4096, 256, 0, stream>>>((float4*)probs, n4);
    vocab_norm_kernel<<<(V + 7) / 8, 256, 0, stream>>>(vocab, vn, V);
    knn_router_kernel<<<(N / M_BLK) * V_SPLIT, 256, 0, stream>>>(emb, vn, candS, candI, V);
    merge_kernel<<<(N + 255) / 256, 256, 0, stream>>>(candS, candI, probs, toks, N, V);
}
